// ReacFormerHybridEncoder_85323820303041
// MI455X (gfx1250) — matrix operand order repaired and checked by reading, not compiled, not run on hardware
//
#include <hip/hip_runtime.h>
#include <hip/hip_bf16.h>

// ---- problem constants ----
#define BB 2
#define NN 64
#define DD 256
#define RR 128
#define KK 16
#define HH 256
#define NE (BB*NN*NN)      // 8192 edges
#define NTILE (NE/16)      // 512 tiles of 16 edges

// ---- WMMA types ----
typedef __attribute__((ext_vector_type(16))) _Float16 v16h;
typedef __attribute__((ext_vector_type(8)))  _Float16 v8h;
typedef __attribute__((ext_vector_type(8)))  float    v8f;

#define LDX 648            // X row stride (f16 elems) for K<=640, multiple of 8
#define LDX2 520           // X row stride for K<=512, multiple of 8

// ============================================================
// WMMA tile GEMM: out[16][256] = act(X[16][Kdim] @ W + bias)
// Block = 512 threads = 16 waves; wave wv owns output columns [16*wv,16*wv+16).
// X:  f16 LDS, row stride ldx (mult of 8), zero-padded to Kp=ceil32(Kdim).
// Wp: f16 global, PACKED fragment order: Wp[(kt*256 + n)*32 + kk], kt=K/32 tile,
//     n=output col, kk=K within tile (zero-padded rows). Each lane's B fragment
//     is two contiguous b128 loads; no LDS staging, no inner barriers.
// outO: f32 LDS, row stride ldo. act: 0=none, 1=silu.
// ============================================================
static __device__ __forceinline__ void gemm_tile(
    const _Float16* __restrict__ X, int ldx, int Kdim,
    const _Float16* __restrict__ Wp, const float* __restrict__ bias,
    float* outO, int ldo, int act)
{
    const int lane = threadIdx.x & 31;
    const int wv   = threadIdx.x >> 5;   // 0..15
    const int m    = lane & 15;
    const int hi   = lane >> 4;          // 0 or 1
    const int n    = wv * 16 + m;        // output column 0..255
    v8f c = {0.f,0.f,0.f,0.f,0.f,0.f,0.f,0.f};
    const int nkt = (Kdim + 31) >> 5;
    __syncthreads();   // X writes visible; previous consumers of outO done
    for (int kt = 0; kt < nkt; ++kt) {
        const int k0 = kt << 5;
        const _Float16* wptr = Wp + ((size_t)(kt*256 + n) << 5) + hi*16;
        if (kt + 1 < nkt)
            __builtin_prefetch(Wp + ((size_t)((kt+1)*256 + n) << 5) + hi*16, 0, 3);
        // A fragment: 16 contiguous halves k0 + 16*hi .. +15 (matches packed B)
        v8h a0 = *(const v8h*)(X + m*ldx + k0 + hi*16);
        v8h a1 = *(const v8h*)(X + m*ldx + k0 + hi*16 + 8);
        // B fragment: 16 contiguous halves from packed weights
        v8h b0 = *(const v8h*)(wptr);
        v8h b1 = *(const v8h*)(wptr + 8);
        v16h a = __builtin_shufflevector(a0, a1, 0,1,2,3,4,5,6,7,8,9,10,11,12,13,14,15);
        v16h b = __builtin_shufflevector(b0, b1, 0,1,2,3,4,5,6,7,8,9,10,11,12,13,14,15);
        c = __builtin_amdgcn_wmma_f32_16x16x32_f16(
                false, a, false, b, (short)0, c, false, false);
    }
    #pragma unroll
    for (int r = 0; r < 8; ++r) {
        float val = c[r] + (bias ? bias[n] : 0.0f);
        if (act) val = val / (1.0f + __expf(-val));   // silu
        outO[(r + 8*hi)*ldo + n] = val;
    }
    __syncthreads();
}

// ============================================================
// K0a: pack weights f32 -> f16 in WMMA B-fragment order
// dst[(kt*256 + n)*32 + kk] = src[(kt*32+kk)*256 + n], zero-padded to Kp
// ============================================================
__global__ void pack_kernel(const float* __restrict__ src,
                            _Float16* __restrict__ dst, int Kdim, int total)
{
    int t = blockIdx.x * blockDim.x + threadIdx.x;
    if (t >= total) return;
    int kk = t & 31;
    int n  = (t >> 5) & 255;
    int kt = t >> 13;          // 256*32 = 8192 per K-tile
    int k  = kt*32 + kk;
    dst[t] = (k < Kdim) ? (_Float16)src[(size_t)k*256 + n] : (_Float16)0.0f;
}

// K0b: plain f32 -> f16 vector conversion (for ts_w2)
__global__ void convert_kernel(const float* __restrict__ src,
                               _Float16* __restrict__ dst, int n)
{
    int t = blockIdx.x * blockDim.x + threadIdx.x;
    if (t < n) dst[t] = (_Float16)src[t];
}

// ============================================================
// K1: per-(b,i) top-16 nearest neighbours (stable ties like lax.top_k)
// ============================================================
__global__ void topk_kernel(const float* __restrict__ dist,
                            const unsigned char* __restrict__ mask,
                            int* __restrict__ tidx, float* __restrict__ tmask)
{
    __shared__ float red[NN];
    int b = blockIdx.x;
    int i = threadIdx.x;
    const float* drow = dist + (b*NN + i)*NN;
    const unsigned char* mrow = mask + (b*NN + i)*NN;
    float mx = -1e30f;
    for (int j = 0; j < NN; ++j) mx = fmaxf(mx, drow[j]);
    red[i] = mx; __syncthreads();
    for (int s = 32; s > 0; s >>= 1) {
        if (i < s) red[i] = fmaxf(red[i], red[i+s]);
        __syncthreads();
    }
    float maxd = fmaxf(red[0], 1.0f) + 1.0f;
    float v[NN];
    for (int j = 0; j < NN; ++j) v[j] = mrow[j] ? drow[j] : maxd;
    unsigned long long used = 0ull;
    for (int k = 0; k < KK; ++k) {
        float best = 1e30f; int bj = 0;
        for (int j = 0; j < NN; ++j) {
            if (!((used >> j) & 1ull) && v[j] < best) { best = v[j]; bj = j; }
        }
        used |= (1ull << bj);
        tidx [(b*NN + i)*KK + k] = bj;
        tmask[(b*NN + i)*KK + k] = mrow[bj] ? 1.f : 0.f;
    }
}

// ============================================================
// K2: edge_base = silu([src|dst|rbf] @ ep_w1 + b1) @ ep_w2 + b2, * access
// ============================================================
__global__ __launch_bounds__(512) void edge_base_kernel(
    const float* __restrict__ node_s, const float* __restrict__ rbf,
    const unsigned char* __restrict__ mask,
    const _Float16* __restrict__ w1, const float* __restrict__ b1,
    const _Float16* __restrict__ w2, const float* __restrict__ b2,
    float* __restrict__ edge_base)
{
    __shared__ __attribute__((aligned(16))) _Float16 X[16*LDX];
    __shared__ __attribute__((aligned(16))) _Float16 X2[16*LDX2];
    __shared__ float O[16*256];
    int e0 = blockIdx.x * 16;
    for (int t = threadIdx.x; t < 16*LDX; t += 512) {
        int r = t / LDX, col = t % LDX;
        int e = e0 + r;
        int b = e >> 12, rem = e & 4095, i = rem >> 6, j = rem & 63;
        float val = 0.f;
        if      (col < 256) val = node_s[(b*NN + i)*DD + col];
        else if (col < 512) val = node_s[(b*NN + j)*DD + (col-256)];
        else if (col < 640) val = rbf[((b*NN + i)*NN + j)*RR + (col-512)];
        X[r*LDX + col] = (_Float16)val;
    }
    gemm_tile(X, LDX, 2*DD+RR, w1, b1, O, 256, 1);
    for (int t = threadIdx.x; t < 16*256; t += 512) {
        int r = t >> 8, col = t & 255;
        X2[r*LDX2 + col] = (_Float16)O[t];
    }
    gemm_tile(X2, LDX2, HH, w2, b2, O, 256, 0);
    for (int t = threadIdx.x; t < 16*256; t += 512) {
        int r = t >> 8, col = t & 255;
        int e = e0 + r;
        int b = e >> 12, rem = e & 4095, i = rem >> 6, j = rem & 63;
        float acc = mask[(b*NN + i)*NN + j] ? 1.f : 0.f;
        edge_base[(size_t)e*DD + col] = O[t] * acc;
    }
}

// ============================================================
// K3: fused per-edge tensor-product + attention
// ============================================================
__global__ __launch_bounds__(512) void pair_kernel(
    const float* __restrict__ node_s, const float* __restrict__ rbf,
    const float* __restrict__ r_hat, const unsigned char* __restrict__ mask,
    const int* __restrict__ tidx, const float* __restrict__ tmaskArr,
    const _Float16* __restrict__ tpw1, const float* __restrict__ tpb1,
    const _Float16* __restrict__ tpw2, const float* __restrict__ tpb2,
    const _Float16* __restrict__ tsw1, const float* __restrict__ tsb1,
    const _Float16* __restrict__ tsw2, const float* __restrict__ tsb2,
    const float* __restrict__ mixlng, const float* __restrict__ mixlnb,
    _Float16* __restrict__ mixin)
{
    __shared__ __attribute__((aligned(16))) _Float16 X2[16*LDX2];
    __shared__ float O[16*256];
    __shared__ float TW[16*256];
    __shared__ float NF[16*256];
    __shared__ float sLog[KK], sPm[KK], sAttn[KK];
    __shared__ int   sIdx[KK];
    __shared__ float yvec[512];
    __shared__ float red[512];

    int e = blockIdx.x;
    int b = e >> 12, rem = e & 4095, i = rem >> 6, j = rem & 63;

    if (threadIdx.x < KK)
        sIdx[threadIdx.x] = tidx[(b*NN + i)*KK + threadIdx.x];
    __syncthreads();

    float rh0 = r_hat[((b*NN + i)*NN + j)*3 + 0];
    float rh1 = r_hat[((b*NN + i)*NN + j)*3 + 1];
    float rh2 = r_hat[((b*NN + i)*NN + j)*3 + 2];

    // X = [legendre(cos) (4) | radial (128)], padded to 160
    for (int t = threadIdx.x; t < 16*160; t += 512) {
        int k = t / 160, col = t % 160;
        int idx = sIdx[k];
        float val = 0.f;
        if (col < 4) {
            const float* rr = r_hat + ((b*NN + i)*NN + idx)*3;
            float cth = rh0*rr[0] + rh1*rr[1] + rh2*rr[2];
            cth = fminf(fmaxf(cth, -1.f + 1e-6f), 1.f - 1e-6f);
            float p0 = 1.f, p1 = cth;
            float p2 = (3.f*cth*p1 - p0) * 0.5f;
            float p3 = (5.f*cth*p2 - 2.f*p1) * (1.f/3.f);
            val = (col == 0) ? p0 : (col == 1) ? p1 : (col == 2) ? p2 : p3;
        } else if (col < 4 + RR) {
            val = rbf[((b*NN + i)*NN + idx)*RR + (col-4)];
        }
        X2[k*LDX2 + col] = (_Float16)val;
    }
    gemm_tile(X2, LDX2, 4 + RR, tpw1, tpb1, O, 256, 1);
    for (int t = threadIdx.x; t < 16*256; t += 512) {
        int k = t >> 8, col = t & 255;
        X2[k*LDX2 + col] = (_Float16)O[t];
    }
    gemm_tile(X2, LDX2, HH, tpw2, tpb2, TW, 256, 0);

    // X2 = [tw | nf] f16; NF = nf f32
    for (int t = threadIdx.x; t < 16*256; t += 512) {
        int k = t >> 8, col = t & 255;
        X2[k*LDX2 + col] = (_Float16)TW[t];
        float nfv = node_s[(b*NN + sIdx[k])*DD + col];
        NF[t] = nfv;
        X2[k*LDX2 + 256 + col] = (_Float16)nfv;
    }
    gemm_tile(X2, LDX2, 2*DD, tsw1, tsb1, O, 256, 1);

    // logits + pair mask
    if (threadIdx.x < KK) {
        int k = threadIdx.x;
        float s = tsb2[0];
        for (int d2 = 0; d2 < 256; ++d2) s += O[k*256 + d2] * (float)tsw2[d2];
        sLog[k] = s;
        float pm = (mask[(b*NN + i)*NN + j] ? 1.f : 0.f)
                 * tmaskArr[(b*NN + i)*KK + k];
        if (sIdx[k] == j) pm = 0.f;
        sPm[k] = pm;
    }
    __syncthreads();
    if (threadIdx.x == 0) {
        float mx = -__builtin_inff(); int any = 0;
        for (int k = 0; k < KK; ++k)
            if (sPm[k] > 0.f) { any = 1; mx = fmaxf(mx, sLog[k]); }
        float ssum = 0.f;
        for (int k = 0; k < KK; ++k) {
            float ev = (sPm[k] > 0.f) ? __expf(sLog[k] - mx) : 0.f;
            sAttn[k] = ev; ssum += ev;
        }
        float inv = (any && ssum > 0.f) ? (1.f / ssum) : 0.f;
        for (int k = 0; k < KK; ++k) sAttn[k] *= inv;
    }
    __syncthreads();

    // t_attn (threads 0..255) and tmax (threads 256..511)
    {
        int tx = threadIdx.x;
        if (tx < 256) {
            float s = 0.f;
            for (int k = 0; k < KK; ++k)
                s += TW[k*256 + tx] * NF[k*256 + tx] * sAttn[k];
            yvec[tx] = s;
        } else {
            int d2 = tx - 256;
            float mxv = -__builtin_inff();
            for (int k = 0; k < KK; ++k)
                if (sPm[k] > 0.f)
                    mxv = fmaxf(mxv, TW[k*256 + d2] * NF[k*256 + d2]);
            yvec[256 + d2] = (mxv == -__builtin_inff()) ? 0.f : mxv;
        }
    }
    __syncthreads();

    // LayerNorm over 512 (mix_ln), store f16 mixin feature
    float v = yvec[threadIdx.x];
    red[threadIdx.x] = v; __syncthreads();
    for (int s = 256; s > 0; s >>= 1) {
        if ((int)threadIdx.x < s) red[threadIdx.x] += red[threadIdx.x + s];
        __syncthreads();
    }
    float mean = red[0] * (1.f/512.f); __syncthreads();
    float dv = v - mean;
    red[threadIdx.x] = dv*dv; __syncthreads();
    for (int s = 256; s > 0; s >>= 1) {
        if ((int)threadIdx.x < s) red[threadIdx.x] += red[threadIdx.x + s];
        __syncthreads();
    }
    float var = red[0] * (1.f/512.f);
    float y = dv * rsqrtf(var + 1e-5f) * mixlng[threadIdx.x] + mixlnb[threadIdx.x];
    mixin[(size_t)e*512 + threadIdx.x] = (_Float16)y;
}

// ============================================================
// K4: ctx MLP + edge-LN + gate -> ef
// ============================================================
__global__ __launch_bounds__(512) void ctx_kernel(
    const _Float16* __restrict__ mixin, const float* __restrict__ edge_base,
    const _Float16* __restrict__ mw1, const float* __restrict__ mb1,
    const _Float16* __restrict__ mw2, const float* __restrict__ mb2,
    const float* __restrict__ elng, const float* __restrict__ elnb,
    const _Float16* __restrict__ gw, const float* __restrict__ gb,
    float* __restrict__ ef)
{
    __shared__ __attribute__((aligned(16))) _Float16 X2[16*LDX2];
    __shared__ float O[16*256];
    __shared__ float E[16*256];
    int e0 = blockIdx.x * 16;
    for (int t = threadIdx.x; t < 16*512; t += 512) {
        int r = t >> 9, col = t & 511;
        X2[r*LDX2 + col] = mixin[(size_t)(e0 + r)*512 + col];
    }
    gemm_tile(X2, LDX2, 2*DD, mw1, mb1, O, 256, 1);
    for (int t = threadIdx.x; t < 16*256; t += 512) {
        int r = t >> 8, col = t & 255;
        X2[r*LDX2 + col] = (_Float16)O[t];
    }
    gemm_tile(X2, LDX2, HH, mw2, mb2, O, 256, 0);
    for (int t = threadIdx.x; t < 16*256; t += 512)
        E[t] = O[t] + edge_base[(size_t)e0*DD + t];
    __syncthreads();
    if (threadIdx.x < 16) {
        int r = threadIdx.x;
        float mean = 0.f;
        for (int d = 0; d < 256; ++d) mean += E[r*256 + d];
        mean *= (1.f/256.f);
        float var = 0.f;
        for (int d = 0; d < 256; ++d) { float dvv = E[r*256 + d] - mean; var += dvv*dvv; }
        var *= (1.f/256.f);
        float inv = rsqrtf(var + 1e-5f);
        for (int d = 0; d < 256; ++d)
            E[r*256 + d] = (E[r*256 + d] - mean) * inv * elng[d] + elnb[d];
    }
    __syncthreads();
    for (int t = threadIdx.x; t < 16*256; t += 512) {
        int r = t >> 8, col = t & 255;
        X2[r*LDX2 + col] = (_Float16)E[t];
    }
    gemm_tile(X2, LDX2, DD, gw, gb, O, 256, 0);
    for (int t = threadIdx.x; t < 16*256; t += 512) {
        float g = 1.f / (1.f + __expf(-O[t]));
        ef[(size_t)e0*DD + t] = g * E[t];
    }
}

// ============================================================
// K5: node_delta = LN(sum_j ef*access) @ node_w + node_b
// ============================================================
__global__ void node_kernel(const float* __restrict__ ef,
                            const unsigned char* __restrict__ mask,
                            const float* __restrict__ nlg, const float* __restrict__ nlb,
                            const float* __restrict__ nw,  const float* __restrict__ nb,
                            float* __restrict__ out)
{
    __shared__ float sln[256];
    __shared__ float red2[256];
    int bi = blockIdx.x;
    int d  = threadIdx.x;
    int b = bi >> 6, i = bi & 63;
    float acc = 0.f;
    for (int j = 0; j < NN; ++j) {
        float a = mask[(b*NN + i)*NN + j] ? 1.f : 0.f;
        acc += ef[((size_t)bi*NN + j)*DD + d] * a;
    }
    red2[d] = acc; __syncthreads();
    for (int st = 128; st > 0; st >>= 1) {
        if (d < st) red2[d] += red2[d + st];
        __syncthreads();
    }
    float mean = red2[0] * (1.f/256.f); __syncthreads();
    float dv = acc - mean;
    red2[d] = dv*dv; __syncthreads();
    for (int st = 128; st > 0; st >>= 1) {
        if (d < st) red2[d] += red2[d + st];
        __syncthreads();
    }
    float var = red2[0] * (1.f/256.f);
    sln[d] = dv * rsqrtf(var + 1e-5f) * nlg[d] + nlb[d];
    __syncthreads();
    float o = nb[d];
    for (int d2 = 0; d2 < 256; ++d2) o += sln[d2] * nw[d2*256 + d];
    out[bi*256 + d] = o;
}

// ============================================================
// K6: bond_graph = sum_{i,j} ef / clip(sum access, 1)
// ============================================================
__global__ void bond_kernel(const float* __restrict__ ef,
                            const unsigned char* __restrict__ mask,
                            float* __restrict__ out)
{
    __shared__ float denomS[BB];
    int t = threadIdx.x;
    int b = t >> 8, d = t & 255;
    if (t < BB) {
        float s = 0.f;
        for (int x = 0; x < NN*NN; ++x) s += mask[t*NN*NN + x] ? 1.f : 0.f;
        denomS[t] = fmaxf(s, 1.f);
    }
    __syncthreads();
    float s = 0.f;
    for (int x = 0; x < NN*NN; ++x)
        s += ef[((size_t)b*NN*NN + x)*DD + d];
    out[BB*NN*DD + b*256 + d] = s / denomS[b];
}

// ============================================================
// Launch
// ============================================================
extern "C" void kernel_launch(void* const* d_in, const int* in_sizes, int n_in,
                              void* d_out, int out_size, void* d_ws, size_t ws_size,
                              hipStream_t stream)
{
    (void)in_sizes; (void)n_in; (void)out_size; (void)ws_size;
    const float* node_s   = (const float*)d_in[0];
    const float* dist     = (const float*)d_in[1];
    const float* rbf      = (const float*)d_in[2];
    const float* r_hat    = (const float*)d_in[3];
    const unsigned char* mask = (const unsigned char*)d_in[4];
    const float* ep_w1 = (const float*)d_in[5];
    const float* ep_b1 = (const float*)d_in[6];
    const float* ep_w2 = (const float*)d_in[7];
    const float* ep_b2 = (const float*)d_in[8];
    const float* tp_w1 = (const float*)d_in[9];
    const float* tp_b1 = (const float*)d_in[10];
    const float* tp_w2 = (const float*)d_in[11];
    const float* tp_b2 = (const float*)d_in[12];
    const float* ts_w1 = (const float*)d_in[13];
    const float* ts_b1 = (const float*)d_in[14];
    const float* ts_w2 = (const float*)d_in[15];
    const float* ts_b2 = (const float*)d_in[16];
    const float* mix_ln_g = (const float*)d_in[17];
    const float* mix_ln_b = (const float*)d_in[18];
    const float* mix_w1 = (const float*)d_in[19];
    const float* mix_b1 = (const float*)d_in[20];
    const float* mix_w2 = (const float*)d_in[21];
    const float* mix_b2 = (const float*)d_in[22];
    const float* gate_w = (const float*)d_in[23];
    const float* gate_b = (const float*)d_in[24];
    const float* node_ln_g = (const float*)d_in[25];
    const float* node_ln_b = (const float*)d_in[26];
    const float* node_w = (const float*)d_in[27];
    const float* node_b = (const float*)d_in[28];
    const float* edge_ln_g = (const float*)d_in[29];
    const float* edge_ln_b = (const float*)d_in[30];
    float* out = (float*)d_out;

    // workspace layout (packed weight buffers sized at Kp = ceil32(K))
    char* base = (char*)d_ws;
    size_t off = 0;
    auto take = [&](size_t bytes) -> char* {
        char* p = base + off;
        off += (bytes + 255) & ~(size_t)255;
        return p;
    };
    auto packed_elems = [](int Kdim) -> size_t {
        return (size_t)(((Kdim + 31) >> 5)) * 256 * 32;
    };
    _Float16* p_ep_w1  = (_Float16*)take(packed_elems(2*DD+RR)*2);
    _Float16* p_ep_w2  = (_Float16*)take(packed_elems(HH)*2);
    _Float16* p_tp_w1  = (_Float16*)take(packed_elems(4+RR)*2);
    _Float16* p_tp_w2  = (_Float16*)take(packed_elems(HH)*2);
    _Float16* p_ts_w1  = (_Float16*)take(packed_elems(2*DD)*2);
    _Float16* h_ts_w2  = (_Float16*)take((size_t)HH*2);
    _Float16* p_mix_w1 = (_Float16*)take(packed_elems(2*DD)*2);
    _Float16* p_mix_w2 = (_Float16*)take(packed_elems(HH)*2);
    _Float16* p_gate_w = (_Float16*)take(packed_elems(DD)*2);
    int*      topk_idx = (int*)take((size_t)BB*NN*KK*4);
    float*    topk_msk = (float*)take((size_t)BB*NN*KK*4);
    float*    edge_bse = (float*)take((size_t)NE*DD*4);
    _Float16* mixin    = (_Float16*)take((size_t)NE*2*DD*2);
    float*    ef       = (float*)take((size_t)NE*DD*4);

    auto pack = [&](const float* s, _Float16* dptr, int Kdim) {
        int total = (int)packed_elems(Kdim);
        pack_kernel<<<(total + 255)/256, 256, 0, stream>>>(s, dptr, Kdim, total);
    };
    pack(ep_w1,  p_ep_w1,  2*DD+RR);
    pack(ep_w2,  p_ep_w2,  HH);
    pack(tp_w1,  p_tp_w1,  4+RR);
    pack(tp_w2,  p_tp_w2,  HH);
    pack(ts_w1,  p_ts_w1,  2*DD);
    pack(mix_w1, p_mix_w1, 2*DD);
    pack(mix_w2, p_mix_w2, HH);
    pack(gate_w, p_gate_w, DD);
    convert_kernel<<<1, 256, 0, stream>>>(ts_w2, h_ts_w2, HH);

    topk_kernel<<<BB, NN, 0, stream>>>(dist, mask, topk_idx, topk_msk);

    edge_base_kernel<<<NTILE, 512, 0, stream>>>(
        node_s, rbf, mask, p_ep_w1, ep_b1, p_ep_w2, ep_b2, edge_bse);

    pair_kernel<<<NE, 512, 0, stream>>>(
        node_s, rbf, r_hat, mask, topk_idx, topk_msk,
        p_tp_w1, tp_b1, p_tp_w2, tp_b2,
        p_ts_w1, ts_b1, h_ts_w2, ts_b2,
        mix_ln_g, mix_ln_b, mixin);

    ctx_kernel<<<NTILE, 512, 0, stream>>>(
        mixin, edge_bse, p_mix_w1, mix_b1, p_mix_w2, mix_b2,
        edge_ln_g, edge_ln_b, p_gate_w, gate_b, ef);

    node_kernel<<<BB*NN, 256, 0, stream>>>(
        ef, mask, node_ln_g, node_ln_b, node_w, node_b, out);

    bond_kernel<<<1, 512, 0, stream>>>(ef, mask, out);
}
